// CKConv_66554813219163
// MI455X (gfx1250) — compile-verified, matrix operand order checked by reading, not run on hardware
//
#include <hip/hip_runtime.h>
#include <hip/hip_bf16.h>

// ---------------------------------------------------------------------------
// CKConv via exact rank-33 factorization of the SIREN-generated kernel:
//   K[o,i,tau] = sum_{r<32} W3[o,i,r]*h[tau,r] + b3[o,i]
// K3 : G[b,i,t,r] = sum_d h_rev[d,r]*x[b,i,t-d]      (WMMA bf16, k = shift d)
// K4 : out[b,o,t] = sum_{i,r} G * W3  +  sum_i b3[o,i]*cumsum(x)[b,i,t]
//      (single WMMA GEMM: k-chunks 0..127 carry (i,r), 128..131 carry cumsum)
// All operands pre-linearized in per-lane WMMA fragment order: hot loops are
// pure {b128 loads -> v_wmma_f32_16x16x32_bf16}.
// ---------------------------------------------------------------------------

typedef __attribute__((ext_vector_type(16))) __bf16 v16bf;
typedef __attribute__((ext_vector_type(8)))  float  v8f;

constexpr int B_   = 32;
constexpr int IC   = 128;
constexpr int OC   = 128;
constexpr int HID_ = 32;
constexpr int L_   = 4096;
constexpr int RP   = 32;            // exact MLP rank (b3 folded via cumsum path)
constexpr int LP   = L_ + 64;       // reversed-x row, zero tail = causality
constexpr int TC   = 128;           // t-chunk so G stays L2-resident
constexpr int NCHUNK = L_ / TC;     // 32
constexpr int NKC  = IC * RP / 32 + 4;  // 128 (i,r) chunks + 4 cumsum chunks = 132
constexpr int NDC  = L_ / 32;       // 128 shift-chunks for basis conv

// workspace layout (bytes)
constexpr size_t OFF_HB = 0;
constexpr size_t SZ_HB  = (size_t)NDC * 2 * 32 * 16 * 2;              // 256 KB
constexpr size_t OFF_WM = OFF_HB + SZ_HB;
constexpr size_t SZ_WM  = (size_t)NKC * 8 * 32 * 16 * 2;              // ~1 MB
constexpr size_t OFF_XR = OFF_WM + SZ_WM;
constexpr size_t SZ_XR  = (size_t)B_ * IC * LP * 2;                   // 34 MB
constexpr size_t OFF_CX = OFF_XR + SZ_XR;
constexpr size_t SZ_CX  = (size_t)B_ * IC * L_ * 2;                   // 33.5 MB
constexpr size_t OFF_G  = OFF_CX + SZ_CX;
constexpr size_t SZ_G   = (size_t)B_ * (TC / 16) * NKC * 32 * 16 * 2; // 34.6 MB

// 16-bit A-fragment (16x32) lane->K mapping per CDNA5 ISA 7.12.2
__device__ __forceinline__ int kmap(int lane, int slot) {
  if (lane < 16) return (slot < 8) ? slot : slot + 8;    // K {0..7, 16..23}
  return (slot < 8) ? slot + 8 : slot + 16;              // K {8..15, 24..31}
}

__device__ __forceinline__ unsigned short f2bf(float f) {
  union { float f; unsigned u; } v; v.f = f;
  unsigned r = v.u + 0x7FFFu + ((v.u >> 16) & 1u);       // round-nearest-even
  return (unsigned short)(r >> 16);
}

// --- K1: SIREN basis, time reversed, B-fragment linear order ----------------
__global__ void k1_basis(const float* __restrict__ w1, const float* __restrict__ b1,
                         const float* __restrict__ w2, const float* __restrict__ b2,
                         unsigned short* __restrict__ hB) {
  int d = blockIdx.x * blockDim.x + threadIdx.x;
  if (d >= L_) return;
  int tau = L_ - 1 - d;                                  // hrev[d] = h[L-1-d]
  float rel = -1.f + 2.f * (float)tau / (float)(L_ - 1);
  float h1[HID_], h2[HID_];
  for (int k = 0; k < HID_; ++k) h1[k] = sinf(30.f * (rel * w1[k] + b1[k]));
  for (int j = 0; j < HID_; ++j) {
    float a = 0.f;
    for (int k = 0; k < HID_; ++k) a += h1[k] * w2[j * HID_ + k];
    h2[j] = sinf(30.f * (a + b2[j]));
  }
  int dc = d >> 5, km = d & 31;
  for (int r = 0; r < RP; ++r) {
    int rt = r >> 4, lr = r & 15;
    int lane, slot;
    if      (km < 8)  { lane = lr;      slot = km;      }
    else if (km < 16) { lane = lr + 16; slot = km - 8;  }
    else if (km < 24) { lane = lr;      slot = km - 8;  }
    else              { lane = lr + 16; slot = km - 16; }
    hB[(((size_t)dc * 2 + rt) * 32 + lane) * 16 + slot] = f2bf(h2[r]);
  }
}

// --- K2a: mixing weights (w3 + b3 tail chunks) in B-fragment order ----------
__global__ void k2_wm(const float* __restrict__ w3, const float* __restrict__ b3,
                      unsigned short* __restrict__ Wm) {
  int gid = blockIdx.x * blockDim.x + threadIdx.x;       // (kc*8+ot)*32+lane
  if (gid >= NKC * 8 * 32) return;
  int lane = gid & 31, ot = (gid >> 5) & 7, kc = gid >> 8;
  int o = ot * 16 + (lane & 15);
  for (int s = 0; s < 16; ++s) {
    int km = kmap(lane, s);
    float v;
    if (kc < 128) {                                      // k = i*32+r, i = kc
      v = w3[(size_t)(o * IC + kc) * HID_ + km];
    } else {                                             // b3 tail: i over 32
      int i = (kc - 128) * 32 + km;
      v = b3[o * IC + i];
    }
    Wm[(size_t)gid * 16 + s] = f2bf(v);
  }
}

// --- K2b: x -> bf16, time-reversed, zero tail -------------------------------
__global__ void k2_xrev(const float* __restrict__ x, unsigned short* __restrict__ xr) {
  size_t idx = (size_t)blockIdx.x * blockDim.x + threadIdx.x;
  if (idx >= (size_t)B_ * IC * LP) return;
  size_t bi = idx / LP; int d = (int)(idx % LP);
  xr[idx] = f2bf(d < L_ ? x[bi * L_ + (L_ - 1 - d)] : 0.f);
}

// --- K2c: causal cumsum of x per (b,i) row, wave32 scan ---------------------
__global__ void k2_cumsum(const float* __restrict__ x, unsigned short* __restrict__ cx) {
  int row = blockIdx.x * 8 + (threadIdx.x >> 5);         // b*IC + i
  int l = threadIdx.x & 31;
  const float* xp = x + (size_t)row * L_;
  unsigned short* cp = cx + (size_t)row * L_;
  float carry = 0.f;
  for (int t0 = 0; t0 < L_; t0 += 32) {
    float v = xp[t0 + l];
    for (int off = 1; off < 32; off <<= 1) {             // inclusive wave scan
      float n = __shfl_up(v, off, 32);
      if (l >= off) v += n;
    }
    v += carry;
    cp[t0 + l] = f2bf(v);
    carry = __shfl(v, 31, 32);
  }
}

// --- K3: basis conv. Wave = one channel i, 16 t-rows, 32 r-cols -------------
__global__ void __launch_bounds__(256)
k3_conv(const unsigned short* __restrict__ xr, const unsigned short* __restrict__ hB,
        unsigned short* __restrict__ G, int chunk) {
  __shared__ float lds[16][8 * RP];                      // 16 t x 256 k
  int b = blockIdx.x, tt = blockIdx.y, ig = blockIdx.z;
  int w = threadIdx.x >> 5, l = threadIdx.x & 31;        // wave32
  int i = ig * 8 + w;
  int t0 = chunk * TC + tt * 16;
  int trow = t0 + (l & 15);
  // x[b,i,t-d] = xrev[b,i,(L-1-t)+d]; fragment-half offset +0/+8
  const unsigned short* xrow =
      xr + ((size_t)b * IC + i) * LP + (size_t)(L_ - 1 - trow) + (l < 16 ? 0 : 8);
  v8f acc0 = {}, acc1 = {};
  int nd = (t0 + 16 + 31) >> 5;                          // causal shift-chunks
  for (int dc = 0; dc < nd; ++dc) {
    const unsigned short* abase = xrow + dc * 32;
    __builtin_prefetch(abase + 64, 0, 3);                // near-scope prefetch
    union { uint4 u[2]; v16bf v; } A, Bf;
    A.u[0] = *(const uint4*)(abase);
    A.u[1] = *(const uint4*)(abase + 16);
    const uint4* hp = (const uint4*)hB + (size_t)dc * 128 + l * 2;
    Bf.u[0] = hp[0];  Bf.u[1] = hp[1];
    acc0 = __builtin_amdgcn_wmma_f32_16x16x32_bf16(false, A.v, false, Bf.v,
                                                   (short)0, acc0, false, false);
    Bf.u[0] = hp[64]; Bf.u[1] = hp[65];
    acc1 = __builtin_amdgcn_wmma_f32_16x16x32_bf16(false, A.v, false, Bf.v,
                                                   (short)0, acc1, false, false);
  }
  // dump f32 C/D tiles (lane l owns rows v + (l<16?0:8), col l&15)
  int rowoff = (l < 16) ? 0 : 8;
  int colbase = w * RP;
  for (int v = 0; v < 8; ++v) {
    lds[v + rowoff][colbase +      (l & 15)] = acc0[v];
    lds[v + rowoff][colbase + 16 + (l & 15)] = acc1[v];
  }
  __syncthreads();
  // repack own 32-col block into bf16 A-fragment, store to G (kc == i)
  int kcg = ig * 8 + w;
  union { unsigned short s[16]; uint4 u[2]; } P;
  int row = l & 15;
  for (int s = 0; s < 16; ++s)
    P.s[s] = f2bf(lds[row][colbase + kmap(l, s)]);
  uint4* gp = (uint4*)(G + ((((size_t)b * 8 + tt) * NKC + kcg) * 32 + l) * 16);
  gp[0] = P.u[0]; gp[1] = P.u[1];
}

// --- K3b: pack cumsum(x) into the 4 tail k-chunks of G ----------------------
__global__ void k3b_cxpack(const unsigned short* __restrict__ cx,
                           unsigned short* __restrict__ G, int chunk) {
  int b = blockIdx.x, tt = blockIdx.y, kq = blockIdx.z;  // kq 0..3
  int l = threadIdx.x & 31;
  int row = chunk * TC + tt * 16 + (l & 15);
  union { unsigned short s[16]; uint4 u[2]; } P;
  for (int s = 0; s < 16; ++s) {
    int i = kq * 32 + kmap(l, s);
    P.s[s] = cx[((size_t)b * IC + i) * L_ + row];
  }
  uint4* gp = (uint4*)(G + ((((size_t)b * 8 + tt) * NKC + (128 + kq)) * 32 + l) * 16);
  gp[0] = P.u[0]; gp[1] = P.u[1];
}

// --- K4: mixing GEMM. Wave = one o-tile x four t-tiles (B reused 4x) --------
__global__ void __launch_bounds__(256)
k4_mix(const unsigned short* __restrict__ G, const unsigned short* __restrict__ Wm,
       const float* __restrict__ bias, float* __restrict__ out, int chunk) {
  int b = blockIdx.x, ttg = blockIdx.y;                  // ttg 0..1 -> 4 t-tiles
  int w = threadIdx.x >> 5, l = threadIdx.x & 31;
  v8f acc[4] = {{}, {}, {}, {}};
  const uint4* gb = (const uint4*)G + (size_t)(b * 8 + ttg * 4) * NKC * 64 + l * 2;
  const uint4* wb = (const uint4*)Wm + ((size_t)w * 32 + l) * 2;
  for (int kc = 0; kc < NKC; ++kc) {
    union { uint4 u[2]; v16bf v; } A, Bf;
    Bf.u[0] = wb[(size_t)kc * 512]; Bf.u[1] = wb[(size_t)kc * 512 + 1];
#pragma unroll
    for (int j = 0; j < 4; ++j) {
      const uint4* ap = gb + (size_t)j * NKC * 64 + (size_t)kc * 64;
      __builtin_prefetch(ap + 128, 0, 3);
      A.u[0] = ap[0]; A.u[1] = ap[1];
      acc[j] = __builtin_amdgcn_wmma_f32_16x16x32_bf16(false, A.v, false, Bf.v,
                                                       (short)0, acc[j], false, false);
    }
  }
  int o = w * 16 + (l & 15);
  float bo = bias[o];
#pragma unroll
  for (int j = 0; j < 4; ++j) {
    int t0 = chunk * TC + (ttg * 4 + j) * 16 + (l < 16 ? 0 : 8);
    float* op = out + ((size_t)b * OC + o) * L_ + t0;
    float4 r0 = { acc[j][0] + bo, acc[j][1] + bo, acc[j][2] + bo, acc[j][3] + bo };
    float4 r1 = { acc[j][4] + bo, acc[j][5] + bo, acc[j][6] + bo, acc[j][7] + bo };
    *(float4*)op = r0; *(float4*)(op + 4) = r1;
  }
}

extern "C" void kernel_launch(void* const* d_in, const int* in_sizes, int n_in,
                              void* d_out, int out_size, void* d_ws, size_t ws_size,
                              hipStream_t stream) {
  (void)in_sizes; (void)n_in; (void)out_size; (void)ws_size;
  const float* x    = (const float*)d_in[0];
  const float* w1   = (const float*)d_in[1];
  const float* b1   = (const float*)d_in[2];
  const float* w2   = (const float*)d_in[3];
  const float* b2   = (const float*)d_in[4];
  const float* w3   = (const float*)d_in[5];
  const float* b3   = (const float*)d_in[6];
  const float* bias = (const float*)d_in[7];
  float* out = (float*)d_out;
  char* ws = (char*)d_ws;
  unsigned short* hB = (unsigned short*)(ws + OFF_HB);
  unsigned short* Wm = (unsigned short*)(ws + OFF_WM);
  unsigned short* xr = (unsigned short*)(ws + OFF_XR);
  unsigned short* cx = (unsigned short*)(ws + OFF_CX);
  unsigned short* G  = (unsigned short*)(ws + OFF_G);

  k1_basis<<<(L_ + 255) / 256, 256, 0, stream>>>(w1, b1, w2, b2, hB);
  k2_wm<<<(NKC * 8 * 32 + 255) / 256, 256, 0, stream>>>(w3, b3, Wm);
  size_t nxr = (size_t)B_ * IC * LP;
  k2_xrev<<<(unsigned)((nxr + 255) / 256), 256, 0, stream>>>(x, xr);
  k2_cumsum<<<(B_ * IC) / 8, 256, 0, stream>>>(x, cx);

  for (int c = 0; c < NCHUNK; ++c) {
    k3_conv  <<<dim3(B_, TC / 16, IC / 8), 256, 0, stream>>>(xr, hB, G, c);
    k3b_cxpack<<<dim3(B_, TC / 16, 4),      32, 0, stream>>>(cx, G, c);
    k4_mix   <<<dim3(B_, 2),               256, 0, stream>>>(G, Wm, bias, out, c);
  }
}
